// AVWDCRNN_2594160246844
// MI455X (gfx1250) — matrix operand order repaired and checked by reading, not compiled, not run on hardware
//
#include <hip/hip_runtime.h>
#include <hip/hip_bf16.h>

// ---------------------------------------------------------------------------
// AVWDCRNN (Chebyshev graph-conv GRU) for MI455X / gfx1250.
// Heavy math through v_wmma_f32_16x16x32_bf16 (bf16 in, f32 accum); A-tiles
// streamed via GLOBAL_LOAD_ASYNC_TO_LDS (ASYNCcnt) with LDS double buffering.
// ---------------------------------------------------------------------------

#define Bb_ 16
#define T_ 24
#define N_ 2048
#define D_ 64
#define L_ 2
#define CIN_ 128              // 2*D
#define KC_ 384               // K * CIN = 3*128
#define BN_ (Bb_ * N_)        // 32768
#define BND_ (Bb_ * N_ * D_)  // 2097152
#define CURSZ_ ((size_t)Bb_ * T_ * N_ * D_)  // 50331648

typedef __attribute__((ext_vector_type(16))) __bf16 v16bf;
typedef __attribute__((ext_vector_type(8)))  float  v8f;
typedef int vint4_t __attribute__((vector_size(16)));   // matches builtin param type

#if defined(__gfx1250__) && __has_builtin(__builtin_amdgcn_global_load_async_to_lds_b128)
#define ASYNC_COPY 1
#else
#define ASYNC_COPY 0
#endif

#if ASYNC_COPY
#if __has_builtin(__builtin_amdgcn_s_wait_asynccnt)
#define ASYNC_WAIT() __builtin_amdgcn_s_wait_asynccnt(0)
#else
#define ASYNC_WAIT() asm volatile("s_wait_asynccnt 0" ::: "memory")
#endif
#else
#define ASYNC_WAIT()
#endif

struct Frag32B { uint4 a, b; };

static __device__ __forceinline__ unsigned short f2bf(float f) {
    union { float f; unsigned int u; } v; v.f = f;
    unsigned int u = v.u + 0x7FFFu + ((v.u >> 16) & 1u);  // RNE
    return (unsigned short)(u >> 16);
}

// A fragment (16x32 bf16, M x K). Lanes 0-15: row M=lane, K chunks [0..7],[16..23];
// lanes 16-31: row M=lane-16, K chunks [8..15],[24..31]. LDS row stride = 40.
static __device__ __forceinline__ v16bf load_fragA(const unsigned short* base, int lane) {
    int r = lane & 15, h = lane >> 4;
    const unsigned short* p = base + r * 40 + h * 8;
    Frag32B f;
    f.a = *(const uint4*)(p);
    f.b = *(const uint4*)(p + 16);
    return __builtin_bit_cast(v16bf, f);
}

// B fragment (32x16 bf16, K x N), LDS stored transposed [col][k], stride 40.
// Lanes 0-15: col=lane, K chunks [0..7],[16..23]; lanes 16-31: col=lane-16, K chunks [8..15],[24..31].
static __device__ __forceinline__ v16bf load_fragB(const unsigned short* base, int lane) {
    int c = lane & 15, h = lane >> 4;
    const unsigned short* p = base + c * 40 + h * 8;
    Frag32B f;
    f.a = *(const uint4*)(p);
    f.b = *(const uint4*)(p + 16);
    return __builtin_bit_cast(v16bf, f);
}

// MODE: 0 = store bf16 (graph gemm -> xcat)
//       1 = +bias, sigmoid, store f32 (gates)
//       2 = +bias, tanh,    store f32 (candidate)
//       3 = 2*acc - I, store bf16 (S2 = 2*A@A - I)
template <int NT, int MODE>
__global__ __launch_bounds__(256) void gemm_bf16_wmma(
    const unsigned short* __restrict__ A, int lda, long long aBatch,
    const unsigned short* __restrict__ Bm, int ldb, long long bBatch,
    void* __restrict__ Cv, int ldc, long long cBatch,
    int Kdim, const float* __restrict__ bias)
{
    constexpr int WN = NT / 32;        // WMMA tiles per wave along N
    __shared__ __align__(16) unsigned short sA[2][128 * 40];
    __shared__ __align__(16) unsigned short sB[2][NT * 40];

    const int tid  = threadIdx.x;
    const int lane = tid & 31;
    const int wave = tid >> 5;
    const int wm   = wave >> 1;        // 0..3, M direction (4 waves * 32 rows)
    const int wn   = wave & 1;         // 0..1, N direction (2 waves * NT/2 cols)
    const int m0   = blockIdx.x * 128;
    const int n0   = blockIdx.y * NT;
    const long long batch = blockIdx.z;

    const unsigned short* Ab = A  + batch * aBatch;
    const unsigned short* Bb = Bm + batch * bBatch;

    // ---- tile loaders (A: straight copy -> async DMA to LDS; B: packed transpose) ----
    auto load_tiles = [&](int kb, int bf) {
        unsigned short* dA = &sA[bf][0];
        unsigned short* dB = &sB[bf][0];
        // A tile: 128 rows x 32 bf16, row-major, LDS stride 40 (512 x 16B chunks)
#pragma unroll
        for (int it = 0; it < 2; ++it) {
            int ch = tid + it * 256;
            int r = ch >> 2, q = ch & 3;
            const unsigned short* src = Ab + (long long)(m0 + r) * lda + kb + q * 8;
#if ASYNC_COPY
            __builtin_amdgcn_global_load_async_to_lds_b128(
                (vint4_t*)src, (vint4_t*)(dA + r * 40 + q * 8), 0, 0);
#else
            *(uint4*)(dA + r * 40 + q * 8) = *(const uint4*)src;
#endif
        }
        // B tile: 32 K-rows x NT cols -> LDS [col][k]; two K-rows packed per b32 store
        constexpr int CG  = NT / 8;      // column groups of 8
        constexpr int NCH = 16 * CG;     // k-pairs * col groups: 256 (NT=128) / 128 (NT=64)
        if (NCH == 256 || tid < NCH) {
            int kp = tid / CG, g = tid % CG;
            const unsigned short* s0 = Bb + (long long)(kb + 2 * kp) * ldb + n0 + g * 8;
            __builtin_prefetch(s0 + 64LL * ldb, 0, 0);   // global_prefetch_b8, 2 K-tiles ahead
            uint4 lo = *(const uint4*)s0;
            uint4 hi = *(const uint4*)(s0 + ldb);
            const unsigned short* pl = (const unsigned short*)&lo;
            const unsigned short* ph = (const unsigned short*)&hi;
#pragma unroll
            for (int e = 0; e < 8; ++e)
                *(unsigned int*)(dB + (g * 8 + e) * 40 + 2 * kp) =
                    (unsigned)pl[e] | ((unsigned)ph[e] << 16);
        }
    };

    v8f acc[2][WN];
#pragma unroll
    for (int mi = 0; mi < 2; ++mi)
#pragma unroll
        for (int ni = 0; ni < WN; ++ni)
            acc[mi][ni] = v8f{0.f, 0.f, 0.f, 0.f, 0.f, 0.f, 0.f, 0.f};

    // prologue: fill buffer 0
    load_tiles(0, 0);
    ASYNC_WAIT();
    __syncthreads();

    int buf = 0;
    for (int kb = 0; kb < Kdim; kb += 32) {
        if (kb + 32 < Kdim) load_tiles(kb + 32, buf ^ 1);   // overlap DMA with WMMA

        v16bf af[2], bfg[WN];
#pragma unroll
        for (int mi = 0; mi < 2; ++mi)
            af[mi] = load_fragA(&sA[buf][0] + (wm * 32 + mi * 16) * 40, lane);
#pragma unroll
        for (int ni = 0; ni < WN; ++ni)
            bfg[ni] = load_fragB(&sB[buf][0] + (wn * (NT / 2) + ni * 16) * 40, lane);
#pragma unroll
        for (int mi = 0; mi < 2; ++mi)
#pragma unroll
            for (int ni = 0; ni < WN; ++ni)
                acc[mi][ni] = __builtin_amdgcn_wmma_f32_16x16x32_bf16(
                    false, af[mi], false, bfg[ni], (short)0, acc[mi][ni], false, false);

        ASYNC_WAIT();
        __syncthreads();
        buf ^= 1;
    }

    // ---- epilogue: C/D layout = VGPR j: M=j (lanes 0-15) / M=j+8 (lanes 16-31), N=lane&15 ----
#pragma unroll
    for (int mi = 0; mi < 2; ++mi) {
#pragma unroll
        for (int ni = 0; ni < WN; ++ni) {
            int rbase   = m0 + wm * 32 + mi * 16 + ((lane >> 4) << 3);
            int c_local = wn * (NT / 2) + ni * 16 + (lane & 15);
            int c       = n0 + c_local;
#pragma unroll
            for (int j = 0; j < 8; ++j) {
                int r = rbase + j;
                float v = acc[mi][ni][j];
                long long off = cBatch * batch + (long long)r * ldc + c;
                if constexpr (MODE == 0) {
                    ((unsigned short*)Cv)[off] = f2bf(v);
                } else if constexpr (MODE == 1) {
                    v += bias[c];
                    ((float*)Cv)[off] = 1.f / (1.f + __expf(-v));
                } else if constexpr (MODE == 2) {
                    v += bias[c];
                    ((float*)Cv)[off] = tanhf(v);
                } else {
                    v = 2.f * v - ((r == c) ? 1.f : 0.f);
                    ((unsigned short*)Cv)[off] = f2bf(v);
                }
            }
        }
    }
}

// ---------------------------- elementwise kernels ----------------------------

__global__ void cast_f32_to_bf16(const float* __restrict__ s, unsigned short* __restrict__ d, int n) {
    int i = blockIdx.x * blockDim.x + threadIdx.x;
    if (i < n) d[i] = f2bf(s[i]);
}

__global__ void copy_f32(const float* __restrict__ s, float* __restrict__ d, int n) {
    int i = blockIdx.x * blockDim.x + threadIdx.x;
    if (i < n) d[i] = s[i];
}

// xcat[bn][0:64] = cur[b,t,n,:], xcat[bn][64:128] = h[bn,:]
__global__ void build_ins(const float* __restrict__ cur, int t,
                          const float* __restrict__ h, unsigned short* __restrict__ xcat) {
    int i = blockIdx.x * blockDim.x + threadIdx.x;
    if (i >= BND_) return;
    int d = i & 63, n = (i >> 6) & (N_ - 1), b = i >> 17;
    int bn = (b << 11) + n;
    float xv = cur[(((long long)(b * T_ + t) * N_ + n) << 6) + d];
    xcat[bn * KC_ + d]      = f2bf(xv);
    xcat[bn * KC_ + 64 + d] = f2bf(h[i]);
}

// xcat[bn][0:64] = cur[b,t,n,:], xcat[bn][64:128] = z[bn,d] * h[bn,d]
__global__ void build_cand(const float* __restrict__ cur, int t,
                           const float* __restrict__ zr, const float* __restrict__ h,
                           unsigned short* __restrict__ xcat) {
    int i = blockIdx.x * blockDim.x + threadIdx.x;
    if (i >= BND_) return;
    int d = i & 63, n = (i >> 6) & (N_ - 1), b = i >> 17;
    int bn = (b << 11) + n;
    float xv = cur[(((long long)(b * T_ + t) * N_ + n) << 6) + d];
    xcat[bn * KC_ + d]      = f2bf(xv);
    xcat[bn * KC_ + 64 + d] = f2bf(zr[bn * CIN_ + d] * h[i]);
}

// h = r*h + (1-r)*hc ; curout[b,t,n,d] = x[b,t,n,d] + h
__global__ void update_h(const float* __restrict__ x, int t,
                         const float* __restrict__ zr, const float* __restrict__ hc,
                         float* __restrict__ h, float* __restrict__ curout) {
    int i = blockIdx.x * blockDim.x + threadIdx.x;
    if (i >= BND_) return;
    int d = i & 63, n = (i >> 6) & (N_ - 1), b = i >> 17;
    int bn = (b << 11) + n;
    float r  = zr[bn * CIN_ + 64 + d];
    float hn = r * h[i] + (1.f - r) * hc[i];
    h[i] = hn;
    long long xi = (((long long)(b * T_ + t) * N_ + n) << 6) + d;
    curout[xi] = x[xi] + hn;
}

// ---------------------------------- host ----------------------------------

extern "C" void kernel_launch(void* const* d_in, const int* in_sizes, int n_in,
                              void* d_out, int out_size, void* d_ws, size_t ws_size,
                              hipStream_t stream) {
    const float* x    = (const float*)d_in[0];
    const float* ist  = (const float*)d_in[1];
    const float* adj  = (const float*)d_in[2];
    const float* Wg   = (const float*)d_in[3];
    const float* bg   = (const float*)d_in[4];
    const float* Wu   = (const float*)d_in[5];
    const float* bu   = (const float*)d_in[6];

    float* out_cur = (float*)d_out;                 // [B,T,N,D]
    float* out_hid = out_cur + CURSZ_;              // [L,B,N,D]

    char* w = (char*)d_ws;
    auto alloc = [&](size_t bytes) {
        char* p = w; w += (bytes + 255) & ~(size_t)255; return p;
    };
    unsigned short* S1   = (unsigned short*)alloc((size_t)N_ * N_ * 2);      // bf16 adjacency (T1)
    unsigned short* S2   = (unsigned short*)alloc((size_t)N_ * N_ * 2);      // bf16 2A^2 - I
    unsigned short* Wgbf = (unsigned short*)alloc((size_t)L_ * KC_ * CIN_ * 2);
    unsigned short* Wubf = (unsigned short*)alloc((size_t)L_ * KC_ * D_ * 2);
    unsigned short* xcat = (unsigned short*)alloc((size_t)BN_ * KC_ * 2);    // [ins | S1*ins | S2*ins]
    float*          zr   = (float*)alloc((size_t)BN_ * CIN_ * 4);
    float*          hc   = (float*)alloc((size_t)BN_ * D_ * 4);
    float*          hbuf = (float*)alloc((size_t)BN_ * D_ * 4);

    const int TPB = 256;
    auto gs = [&](int n) { return dim3((n + TPB - 1) / TPB); };

    // One-time casts to bf16
    cast_f32_to_bf16<<<gs(N_ * N_), TPB, 0, stream>>>(adj, S1, N_ * N_);
    cast_f32_to_bf16<<<gs(L_ * KC_ * CIN_), TPB, 0, stream>>>(Wg, Wgbf, L_ * KC_ * CIN_);
    cast_f32_to_bf16<<<gs(L_ * KC_ * D_), TPB, 0, stream>>>(Wu, Wubf, L_ * KC_ * D_);

    // S2 = 2*A@A - I  (2048x2048x2048 bf16 WMMA GEMM)
    gemm_bf16_wmma<128, 3><<<dim3(N_ / 128, N_ / 128, 1), 256, 0, stream>>>(
        S1, N_, 0, S1, N_, 0, (void*)S2, N_, 0, N_, nullptr);

    const long long bStride = (long long)N_ * KC_;  // per-batch row stride in xcat (elements)

    for (int l = 0; l < L_; ++l) {
        copy_f32<<<gs(BND_), TPB, 0, stream>>>(ist + (size_t)l * BND_, hbuf, BND_);
        const float* curin = (l == 0) ? x : out_cur;

        for (int t = 0; t < T_; ++t) {
            // ---- gate path ----
            build_ins<<<gs(BND_), TPB, 0, stream>>>(curin, t, hbuf, xcat);
            // xg1 = S1 @ ins (per batch), xg2 = S2 @ ins -> xcat cols 128/256
            gemm_bf16_wmma<128, 0><<<dim3(N_ / 128, 1, Bb_), 256, 0, stream>>>(
                S1, N_, 0, xcat, KC_, bStride, (void*)(xcat + 128), KC_, bStride, N_, nullptr);
            gemm_bf16_wmma<128, 0><<<dim3(N_ / 128, 1, Bb_), 256, 0, stream>>>(
                S2, N_, 0, xcat, KC_, bStride, (void*)(xcat + 256), KC_, bStride, N_, nullptr);
            // zr = sigmoid(xcat @ Wg[l] + bg[l])   (32768 x 384 x 128)
            gemm_bf16_wmma<128, 1><<<dim3(BN_ / 128, 1, 1), 256, 0, stream>>>(
                xcat, KC_, 0, Wgbf + (size_t)l * KC_ * CIN_, CIN_, 0,
                (void*)zr, CIN_, 0, KC_, bg + l * CIN_);

            // ---- candidate path ----
            build_cand<<<gs(BND_), TPB, 0, stream>>>(curin, t, zr, hbuf, xcat);
            gemm_bf16_wmma<128, 0><<<dim3(N_ / 128, 1, Bb_), 256, 0, stream>>>(
                S1, N_, 0, xcat, KC_, bStride, (void*)(xcat + 128), KC_, bStride, N_, nullptr);
            gemm_bf16_wmma<128, 0><<<dim3(N_ / 128, 1, Bb_), 256, 0, stream>>>(
                S2, N_, 0, xcat, KC_, bStride, (void*)(xcat + 256), KC_, bStride, N_, nullptr);
            // hc = tanh(xcat @ Wu[l] + bu[l])   (32768 x 384 x 64)
            gemm_bf16_wmma<64, 2><<<dim3(BN_ / 128, 1, 1), 256, 0, stream>>>(
                xcat, KC_, 0, Wubf + (size_t)l * KC_ * D_, D_, 0,
                (void*)hc, D_, 0, KC_, bu + l * D_);

            // h = r*h + (1-r)*hc ; current[t] = x[t] + h  (in place into d_out)
            update_h<<<gs(BND_), TPB, 0, stream>>>(x, t, zr, hc, hbuf, out_cur);
        }
        copy_f32<<<gs(BND_), TPB, 0, stream>>>(hbuf, out_hid + (size_t)l * BND_, BND_);
    }
}
